// CMFMLoss_28862180229715
// MI455X (gfx1250) — matrix operand order repaired and checked by reading, not compiled, not run on hardware
//
#include <hip/hip_runtime.h>
#include <hip/hip_bf16.h>

// Problem constants (B,T,D fixed by the reference).
#define NB 64
#define NT 1024
#define ND 256
#define NK (NT * ND)          // 262144 : K extent of the Gram GEMM
#define KWG 1024              // K-slab per workgroup in pass 2
constexpr float C_ALPHA = 2.0f, C_BETA = 2.0f, C_GAMMA = 1.0f, C_EPS = 1e-8f;

typedef __attribute__((ext_vector_type(16))) __bf16 v16bf;
typedef __attribute__((ext_vector_type(8)))  __bf16 v8bf;
typedef __attribute__((ext_vector_type(8)))  float  v8f;
typedef __attribute__((ext_vector_type(4)))  unsigned int v4u;

union Frag { v4u u[2]; v16bf v; };

// ---------------- Pass 0: zero the 64x64 accumulator ----------------
__global__ void k_zero(float* __restrict__ g) {
    g[blockIdx.x * 256 + threadIdx.x] = 0.0f;   // grid 16 x block 256 = 4096 floats
}

// ---------------- Pass 1: normalize rows, convert to bf16 ----------------
// One wave per (b,t) row of 256 floats; lane handles 8 contiguous elements.
__global__ void k_normalize(const float* __restrict__ fv, const float* __restrict__ fa,
                            __bf16* __restrict__ vn, __bf16* __restrict__ an) {
    const int wave = threadIdx.x >> 5;
    const int lane = threadIdx.x & 31;
    const size_t row = (size_t)blockIdx.x * 8 + wave;          // 0..65535
    const float* src = blockIdx.y ? fa : fv;
    __bf16*      dst = blockIdx.y ? an : vn;

    const float* p = src + row * ND + lane * 8;
    float4 x0 = *(const float4*)p;
    float4 x1 = *(const float4*)(p + 4);

    float s = x0.x*x0.x + x0.y*x0.y + x0.z*x0.z + x0.w*x0.w
            + x1.x*x1.x + x1.y*x1.y + x1.z*x1.z + x1.w*x1.w;
    #pragma unroll
    for (int m = 16; m >= 1; m >>= 1) s += __shfl_xor(s, m, 32);

    const float rinv = 1.0f / fmaxf(sqrtf(s), C_EPS);          // x / max(||x||, eps)

    v8bf o;
    o[0] = (__bf16)(x0.x * rinv); o[1] = (__bf16)(x0.y * rinv);
    o[2] = (__bf16)(x0.z * rinv); o[3] = (__bf16)(x0.w * rinv);
    o[4] = (__bf16)(x1.x * rinv); o[5] = (__bf16)(x1.y * rinv);
    o[6] = (__bf16)(x1.z * rinv); o[7] = (__bf16)(x1.w * rinv);
    *(v8bf*)(dst + row * ND + lane * 8) = o;
}

// ---------------- Pass 2: G += vn(64xK) * an(64xK)^T via WMMA bf16 ----------------
// 8 waves/WG: wave (mi = w&3, njp = w>>2) owns tiles (mi, 2*njp) and (mi, 2*njp+1).
// Each workgroup reduces a KWG-wide K-slab and atomically folds into G.
__global__ void k_gemm(const __bf16* __restrict__ vn, const __bf16* __restrict__ an,
                       float* __restrict__ G) {
    const int tid   = threadIdx.x;
    const int wave  = tid >> 5;
    const int lane  = tid & 31;
    const int r     = lane & 15;        // row-in-tile / column-in-tile
    const int h     = lane >> 4;        // lane half selects K sub-range
    const int iBase = (wave & 3) * 16;
    const int jBase = (wave >> 2) * 32;
    const int kbase = blockIdx.x * KWG;

    // A fragment (16x32 bf16): lane(0-15)=row, VGPR0-3 <- K[h*8..h*8+7], VGPR4-7 <- K[16+h*8..]
    const __bf16* aRow  = vn + (size_t)(iBase + r) * NK + h * 8;
    // B fragment (32x16 bf16): lane%16 = column j, VGPR0-3 <- K[h*8..h*8+7], VGPR4-7 <- K[16+h*8..]
    const __bf16* bCol0 = an + (size_t)(jBase +      r) * NK + h * 8;
    const __bf16* bCol1 = an + (size_t)(jBase + 16 + r) * NK + h * 8;

    v8f c0 = {}; v8f c1 = {};
    for (int kb = kbase; kb < kbase + KWG; kb += 32) {
        Frag A, B0, B1;
        A.u[0]  = *(const v4u*)(aRow  + kb);
        A.u[1]  = *(const v4u*)(aRow  + kb + 16);
        B0.u[0] = *(const v4u*)(bCol0 + kb);
        B0.u[1] = *(const v4u*)(bCol0 + kb + 16);
        B1.u[0] = *(const v4u*)(bCol1 + kb);
        B1.u[1] = *(const v4u*)(bCol1 + kb + 16);
        c0 = __builtin_amdgcn_wmma_f32_16x16x32_bf16(false, A.v, false, B0.v,
                                                     (short)0, c0, false, false);
        c1 = __builtin_amdgcn_wmma_f32_16x16x32_bf16(false, A.v, false, B1.v,
                                                     (short)0, c1, false, false);
    }

    // C layout: VGPR v, lanes 0-15 -> (M=v, N=lane); lanes 16-31 -> (M=v+8, N=lane-16)
    const int i0 = iBase + h * 8;
    #pragma unroll
    for (int v = 0; v < 8; ++v) {
        atomicAdd(&G[(i0 + v) * NB + jBase +      r], c0[v]);
        atomicAdd(&G[(i0 + v) * NB + jBase + 16 + r], c1[v]);
    }
}

// ---------------- Pass 3: fold G + labels into the scalar loss ----------------
__global__ void k_finalize(const float* __restrict__ G, const int* __restrict__ labels,
                           float* __restrict__ out) {
    __shared__ float sPos[NB], sNeg[NB], sCross[NB];
    __shared__ int   sP[NB];
    const int i = threadIdx.x;

    const float diag = G[i * NB + i];
    float rs = 0.0f;
    for (int j = 0; j < NB; ++j) rs += G[i * NB + j];

    const int isPos = (labels[i] == 0);
    sPos[i]   = isPos ? (C_ALPHA * ((float)NT - diag)) : 0.0f;  // alpha * sum_t (1-cos)
    sNeg[i]   = isPos ? 0.0f : (C_BETA * diag);                 // beta  * sum_t cos
    sCross[i] = rs - diag;                                      // off-diagonal row sum
    sP[i]     = isPos;
    __syncthreads();

    if (i == 0) {
        float lp = 0.f, ln = 0.f, cx = 0.f; int np = 0;
        for (int j = 0; j < NB; ++j) { lp += sPos[j]; ln += sNeg[j]; cx += sCross[j]; np += sP[j]; }
        ln += C_GAMMA * cx / (float)NT;
        const float cntPos = (float)np * (float)NT;
        const float cntNeg = (float)(NB - np) * (float)NT + (float)(NB * (NB - 1));
        float loss = 0.0f;
        if (cntPos > 0.0f) loss += lp / fmaxf(cntPos, 1.0f);
        if (cntNeg > 0.0f) loss += ln / fmaxf(cntNeg, 1.0f);
        out[0] = loss;
    }
}

extern "C" void kernel_launch(void* const* d_in, const int* in_sizes, int n_in,
                              void* d_out, int out_size, void* d_ws, size_t ws_size,
                              hipStream_t stream) {
    const float* fv     = (const float*)d_in[0];
    const float* fa     = (const float*)d_in[1];
    const int*   labels = (const int*)d_in[2];

    char* ws = (char*)d_ws;
    const size_t elems = (size_t)NB * NT * ND;          // 16,777,216
    __bf16* vn = (__bf16*)ws;                           // 32 MB
    __bf16* an = (__bf16*)(ws + elems * 2);             // 32 MB
    float*  G  = (float*)(ws + elems * 4);              // 16 KB

    k_zero     <<<16, 256, 0, stream>>>(G);
    k_normalize<<<dim3((NB * NT) / 8, 2), 256, 0, stream>>>(fv, fa, vn, an);
    k_gemm     <<<NK / KWG, 256, 0, stream>>>(vn, an, G);
    k_finalize <<<1, NB, 0, stream>>>(G, labels, (float*)d_out);
}
